// FactEncoder_87316685128430
// MI455X (gfx1250) — matrix operand order repaired and checked by reading, not compiled, not run on hardware
//
#include <hip/hip_runtime.h>

typedef __attribute__((ext_vector_type(16))) __bf16 v16bf;
typedef __attribute__((ext_vector_type(8)))  float  v8f;

#define NSEQ   320     // B*F sequences
#define TT     128     // time steps
#define DD     300     // input dim
#define KXPAD  320     // input K padded to multiple of 32
#define HH     256     // hidden
#define KTOT   576     // KXPAD + HH
#define NKT    18      // K tiles of 32
#define NNT    64      // N tiles of 16 (4H = 1024)
#define FRAG_U16 512   // one 32x16 bf16 B fragment: 32 lanes * 16 u16

__device__ __forceinline__ unsigned short f2bf(float f) {
  unsigned u = __float_as_uint(f);
  u += 0x7fffu + ((u >> 16) & 1u);          // round-to-nearest-even
  return (unsigned short)(u >> 16);
}
__device__ __forceinline__ float sigf(float x) {
  return 1.0f / (1.0f + __expf(-x));
}
__device__ __forceinline__ float tanhfast(float x) {
  float e = __expf(2.0f * x);
  return (e - 1.0f) / (e + 1.0f);
}

// ---------------------------------------------------------------------------
// Prepack W (300x1024, K-padded to 320) and U (256x1024) for both directions
// into bf16 fragments in the exact wave32 WMMA B-matrix register order:
//   lane 0-15 : N = lane,    K slots idx 0..7 -> K = 32*kt + idx, idx 8..15 -> K = 32*kt + 8 + idx
//   lane 16-31: N = lane-16, K slots idx 0..7 -> K = 32*kt + 8 + idx, idx 8..15 -> K = 32*kt + 16 + idx
// Fragment index = dir*NKT*NNT + kt*NNT + nt ; each fragment = 512 u16.
// ---------------------------------------------------------------------------
__global__ void pack_weights_kernel(const float* __restrict__ Wf,
                                    const float* __restrict__ Uf,
                                    const float* __restrict__ Wb,
                                    const float* __restrict__ Ub,
                                    unsigned short* __restrict__ Bpack) {
  int tid = blockIdx.x * blockDim.x + threadIdx.x;   // one thread per (frag, lane)
  const int total = 2 * NKT * NNT * 32;
  if (tid >= total) return;
  int lane = tid & 31;
  int frag = tid >> 5;
  int nt   = frag % NNT;
  int kt   = (frag / NNT) % NKT;
  int dir  = frag / (NKT * NNT);
  const float* W = dir ? Wb : Wf;
  const float* U = dir ? Ub : Uf;
  int n = nt * 16 + (lane & 15);
  unsigned short* dst = Bpack + (size_t)frag * FRAG_U16 + lane * 16;
#pragma unroll
  for (int idx = 0; idx < 16; ++idx) {
    int kl = (idx & 7) + ((lane >> 4) << 3) + ((idx >> 3) << 4);   // K within tile
    int k  = kt * 32 + kl;               // K within [W ; U]
    float v = 0.0f;
    if (k < KXPAD) {
      if (k < DD) v = W[(size_t)k * 1024 + n];
    } else {
      v = U[(size_t)(k - KXPAD) * 1024 + n];
    }
    dst[idx] = f2bf(v);
  }
}

// ---------------------------------------------------------------------------
// Fused bidirectional-LSTM recurrence. One WG = 16 sequences x 1 direction.
// 8 waves; wave w owns N-tiles {2w,2w+1} inside each gate block (i,f,g,o),
// i.e. hidden columns [32w, 32w+32). c/h state stays in registers.
// Weights are re-read from the L2-resident pack every step. A laundered
// 32-bit zero offset makes the addresses loop-variant (no LICM hoist+spill)
// while keeping Bpack as the address root so AS-inference still emits
// global_load_b128 (LOADcnt only), not flat_load (LOADcnt+DScnt).
// ---------------------------------------------------------------------------
__launch_bounds__(256, 1)
__global__ void lstm_recurrent_kernel(const float* __restrict__ facts,
                                      const unsigned char* __restrict__ mask,
                                      const unsigned short* __restrict__ Bpack,
                                      const float* __restrict__ bias_f,
                                      const float* __restrict__ bias_b,
                                      float* __restrict__ out) {
  const int dir     = blockIdx.x & 1;
  const int seqBase = (blockIdx.x >> 1) * 16;
  const int tid  = threadIdx.x;
  const int lane = tid & 31;
  const int wave = tid >> 5;

  // A matrix [16 x 576] in bf16, stored fragment-swizzled per K-tile:
  // lA[kt*512 + lane*16 + idx]
  __shared__ __align__(32) unsigned short lA[NKT * FRAG_U16];   // 18 KB
  __shared__ __align__(8)  unsigned char  smask[16];

  // zero the h-region (kt 10..17); the x-writer handles x padding each step
  for (int i = tid; i < 8 * FRAG_U16; i += 256) lA[10 * FRAG_U16 + i] = 0;

  // wave's 8 N-tiles: {i0,i1,f0,f1,g0,g1,o0,o1}
  int tiles[8];
#pragma unroll
  for (int g = 0; g < 4; ++g) {
    tiles[2 * g + 0] = 16 * g + 2 * wave + 0;
    tiles[2 * g + 1] = 16 * g + 2 * wave + 1;
  }
  const float* bptr = dir ? bias_b : bias_f;
  float biasv[8];
#pragma unroll
  for (int u = 0; u < 8; ++u) biasv[u] = bptr[tiles[u] * 16 + (lane & 15)];

  v8f c_st[2], h_st[2];
#pragma unroll
  for (int x = 0; x < 2; ++x)
#pragma unroll
    for (int v = 0; v < 8; ++v) { c_st[x][v] = 0.0f; h_st[x][v] = 0.0f; }

  const size_t dirOff = (size_t)dir * NKT * NNT * FRAG_U16;
  const int row = (lane >> 4) * 8;           // first of 8 M rows this lane holds

  for (int ts = 0; ts < TT; ++ts) {
    const int t = dir ? (TT - 1 - ts) : ts;

    // ---- stage mask + x_t (bf16, fragment-swizzled) into LDS ----
    if (tid < 16) smask[tid] = mask[(size_t)(seqBase + tid) * TT + t];
    for (int e = tid; e < 16 * KXPAD; e += 256) {
      int m = e / KXPAD;
      int k = e - m * KXPAD;
      float xv = (k < DD)
          ? facts[((size_t)(seqBase + m) * TT + t) * DD + k] : 0.0f;
      int kt  = k >> 5;
      int kl  = k & 31;
      int grp = (kl >> 3) & 1;
      int idx = (kl & 7) + ((kl >> 4) << 3);
      lA[kt * FRAG_U16 + (m + (grp << 4)) * 16 + idx] = f2bf(xv);
    }
    __syncthreads();

    unsigned long long m64 =
        *(const unsigned long long*)(smask + ((lane >> 4) << 3));

    // Laundered zero offset: defeats LICM (which previously hoisted all
    // 144 B fragments and spilled them to scratch) while preserving the
    // global address-space provenance of Bpack.
    unsigned int lz = 0;
    asm volatile("" : "+s"(lz));
    const unsigned short* bbase = Bpack + dirOff + lz;

    // ---- z = [x_t | h] @ [W;U] + b  via 144 WMMAs ----
    v8f acc[8];
#pragma unroll
    for (int u = 0; u < 8; ++u)
#pragma unroll
      for (int v = 0; v < 8; ++v) acc[u][v] = biasv[u];

    for (int kt = 0; kt < NKT; ++kt) {
      v16bf a = *(const v16bf*)(&lA[kt * FRAG_U16 + lane * 16]);
      const unsigned short* brow = bbase + (size_t)kt * NNT * FRAG_U16;
#pragma unroll
      for (int u = 0; u < 8; ++u) {
        v16bf b = *(const v16bf*)(brow + (size_t)tiles[u] * FRAG_U16 + lane * 16);
        acc[u] = __builtin_amdgcn_wmma_f32_16x16x32_bf16(
            false, a, false, b, (short)0, acc[u], false, false);
      }
    }

    // ---- gates, state update (masked), output + h re-staging ----
#pragma unroll
    for (int tix = 0; tix < 2; ++tix) {
#pragma unroll
      for (int v = 0; v < 8; ++v) {
        float zi = acc[0 + tix][v];
        float zf = acc[2 + tix][v];
        float zg = acc[4 + tix][v];
        float zo = acc[6 + tix][v];
        float cn = sigf(zf) * c_st[tix][v] + sigf(zi) * tanhfast(zg);
        float hn = sigf(zo) * tanhfast(cn);
        bool mm = ((m64 >> (8 * v)) & 0xffull) != 0ull;
        if (mm) { c_st[tix][v] = cn; h_st[tix][v] = hn; }

        int m   = row + v;                             // sequence row 0..15
        int col = 32 * wave + 16 * tix + (lane & 15);  // hidden col 0..255
        out[(((size_t)(seqBase + m) * TT) + t) * 512 + dir * 256 + col] =
            h_st[tix][v];

        // restage h (bf16) into A fragment layout at K = KXPAD + col
        int kl  = 16 * tix + (lane & 15);   // K within tile (kt = 10+wave)
        int grp = (kl >> 3) & 1;
        int idx = (kl & 7) + ((kl >> 4) << 3);
        lA[(10 + wave) * FRAG_U16 + (m + (grp << 4)) * 16 + idx] =
            f2bf(h_st[tix][v]);
      }
    }
    __syncthreads();
  }
}

extern "C" void kernel_launch(void* const* d_in, const int* in_sizes, int n_in,
                              void* d_out, int out_size, void* d_ws, size_t ws_size,
                              hipStream_t stream) {
  const float*         facts = (const float*)d_in[0];
  const unsigned char* mask  = (const unsigned char*)d_in[1];
  const float*         W_fwd = (const float*)d_in[2];
  const float*         U_fwd = (const float*)d_in[3];
  const float*         b_fwd = (const float*)d_in[4];
  const float*         W_bwd = (const float*)d_in[5];
  const float*         U_bwd = (const float*)d_in[6];
  const float*         b_bwd = (const float*)d_in[7];
  float* out = (float*)d_out;

  unsigned short* Bpack = (unsigned short*)d_ws;  // 2*18*64*512*2B = 2.25 MB

  const int packThreads = 2 * NKT * NNT * 32;     // 73728
  pack_weights_kernel<<<(packThreads + 255) / 256, 256, 0, stream>>>(
      W_fwd, U_fwd, W_bwd, U_bwd, Bpack);

  lstm_recurrent_kernel<<<(NSEQ / 16) * 2, 256, 0, stream>>>(
      facts, mask, Bpack, b_fwd, b_bwd, out);
}
